// DeepseekV2MoE_72481868087636
// MI455X (gfx1250) — matrix operand order repaired and checked by reading, not compiled, not run on hardware
//
#include <hip/hip_runtime.h>
#include <hip/hip_bf16.h>

// ---------------- problem constants (DeepseekV2 MoE reference) ----------------
#define H_DIM   1024
#define E_NUM   16
#define I_DIM   512
#define IS_DIM  1024
#define T_TOK   2048
#define SCALE_F 1.0f
#define MT32    (T_TOK / 32)   // 64  (32-row M-blocks)
#define BSTRIDE 20             // dwords per column in packed B slab (16 kp + pad)

typedef __attribute__((ext_vector_type(16))) __bf16 bf16x16;
typedef __attribute__((ext_vector_type(2)))  __bf16 bf16x2;
typedef __attribute__((ext_vector_type(8)))  float  f32x8;

// ---------------------------------------------------------------------------
// WMMA helper: D = A(16x32 bf16) * B(32x16 bf16) + C(16x16 f32)
// ---------------------------------------------------------------------------
__device__ __forceinline__ f32x8 wmma_bf16(bf16x16 a, bf16x16 b, f32x8 c) {
  return __builtin_amdgcn_wmma_f32_16x16x32_bf16(
      /*neg_a=*/false, a, /*neg_b=*/false, b,
      /*c_mod=*/(short)0, c, /*reuse_a=*/false, /*reuse_b=*/false);
}

// A-fragment (16x32, 16-bit) from LDS per CDNA5 ISA layout:
//  lane = hi*16 + m ; VGPR j<4 : K = hi*8 + 2j,2j+1 ; j>=4 : K = 16 + hi*8 + ...
// (pairs are dword-contiguous -> compiler emits 2x ds_load_b128)
__device__ __forceinline__ bf16x16 load_a_lds(const __bf16* __restrict__ row, int hi) {
  bf16x16 a;
#pragma unroll
  for (int j = 0; j < 8; ++j) {
    const int kb = ((j < 4) ? (2 * j) : (16 + 2 * (j - 4))) + hi * 8;
    a[2 * j]     = row[kb];
    a[2 * j + 1] = row[kb + 1];
  }
  return a;
}

// pack two fp32 into one dword: low = bf16(lo) (even k), high = bf16(hi) (odd k)
// vector-init of 2 fptruncs -> single v_cvt_pk_bf16_f32
__device__ __forceinline__ unsigned pack_bf16(float lo, float hi) {
  bf16x2 v = { (__bf16)lo, (__bf16)hi };
  return __builtin_bit_cast(unsigned, v);
}

// B slab layout, per column n (BSTRIDE dwords): 16 data dwords as 4 quads of 4,
// quad index XOR-swizzled by ((n>>4)&3) so staging stores are bank-conflict-free
// while fragment reads stay 2x contiguous aligned ds_load_b128.
__device__ __forceinline__ int b_off(int n, int kp) {
  const int q = (kp >> 2) ^ ((n >> 4) & 3);
  return n * BSTRIDE + (q << 2) + (kp & 3);
}

// Stage a 32(K) x NCOLS(N) fp32 tile into wave-private LDS, k-pair packed:
//  slab[n][kp] = {bf16(B[2kp][n]), bf16(B[2kp+1][n])}
// Coalesced float4 global loads; no barriers needed (wave-private, DS in-order).
template<int NCOLS>
__device__ __forceinline__ void stage_b(unsigned* __restrict__ sb,
                                        const float* __restrict__ B,  // &B[kk*ldb + nb]
                                        int ldb, int lane) {
  const int hi  = lane >> 4;
  const int nl4 = (lane & 15) * 4;
#pragma unroll
  for (int i = 0; i < NCOLS / 8; ++i) {
    const int kp = hi + 2 * (i & 7);
    const int n4 = nl4 + (i >> 3) * 64;
    const float4 r0 = *(const float4*)(B + (size_t)(2 * kp)     * ldb + n4);
    const float4 r1 = *(const float4*)(B + (size_t)(2 * kp + 1) * ldb + n4);
    sb[b_off(n4 + 0, kp)] = pack_bf16(r0.x, r1.x);
    sb[b_off(n4 + 1, kp)] = pack_bf16(r0.y, r1.y);
    sb[b_off(n4 + 2, kp)] = pack_bf16(r0.z, r1.z);
    sb[b_off(n4 + 3, kp)] = pack_bf16(r0.w, r1.w);
  }
}

// B-fragment from packed slab: two swizzled quads -> 2x ds_load_b128
__device__ __forceinline__ bf16x16 load_b_lds(const unsigned* __restrict__ sb,
                                              int nloc, int hi) {
  const int s = (nloc >> 4) & 3;
  const unsigned* base = sb + nloc * BSTRIDE;
  const unsigned* p0 = base + ((((hi)         ^ s)) << 2);
  const unsigned* p1 = base + ((((2 + hi)     ^ s)) << 2);
  union { unsigned u[8]; bf16x16 b; } r;
#pragma unroll
  for (int j = 0; j < 4; ++j) r.u[j] = p0[j];
#pragma unroll
  for (int j = 0; j < 4; ++j) r.u[4 + j] = p1[j];
  return r.b;
}

__device__ __forceinline__ float silu_f(float g) {
  return g / (1.0f + __expf(-g));
}

// stage one fp32 row chunk into bf16 LDS: b128 load -> 2x cvt_pk -> b64 store
__device__ __forceinline__ void stage4(__bf16* __restrict__ dst,
                                       const float* __restrict__ src) {
  const float4 v = *(const float4*)src;
  uint2 u;
  u.x = pack_bf16(v.x, v.y);
  u.y = pack_bf16(v.z, v.w);
  *(uint2*)dst = u;
}

// ---------------------------------------------------------------------------
// 0) zero per-expert counters
// ---------------------------------------------------------------------------
__global__ void moe_init(int* __restrict__ cnt) {
  if (threadIdx.x < E_NUM) cnt[threadIdx.x] = 0;
}

// ---------------------------------------------------------------------------
// 1) router: logits -> softmax -> top-2 -> per-expert dispatch lists
// ---------------------------------------------------------------------------
__global__ __launch_bounds__(256)
void moe_router(const float* __restrict__ x, const float* __restrict__ gw,
                int* __restrict__ cnt, int* __restrict__ tok,
                float* __restrict__ wgt) {
  const int wave = threadIdx.x >> 5;
  const int lane = threadIdx.x & 31;
  const int t    = blockIdx.x * 8 + wave;

  float acc[E_NUM];
#pragma unroll
  for (int e = 0; e < E_NUM; ++e) acc[e] = 0.0f;

  const float* xr = x + (size_t)t * H_DIM;
  for (int h = lane; h < H_DIM; h += 32) {
    const float xv = xr[h];
#pragma unroll
    for (int e = 0; e < E_NUM; ++e) acc[e] += xv * gw[e * H_DIM + h];
  }
#pragma unroll
  for (int e = 0; e < E_NUM; ++e) {
#pragma unroll
    for (int off = 16; off > 0; off >>= 1)
      acc[e] += __shfl_xor(acc[e], off, 32);
  }

  float mx = acc[0];
#pragma unroll
  for (int e = 1; e < E_NUM; ++e) mx = fmaxf(mx, acc[e]);
  float p[E_NUM], se = 0.0f;
#pragma unroll
  for (int e = 0; e < E_NUM; ++e) { p[e] = __expf(acc[e] - mx); se += p[e]; }
  const float inv = 1.0f / se;

  int   i1 = 0, i2 = -1;
  float v1 = -1.0f, v2 = -1.0f;
#pragma unroll
  for (int e = 0; e < E_NUM; ++e) {
    const float pe = p[e];
    if (pe > v1)      { v2 = v1; i2 = i1; v1 = pe; i1 = e; }
    else if (pe > v2) { v2 = pe; i2 = e; }
  }

  if (lane == 0) {
    int s1 = atomicAdd(&cnt[i1], 1);
    tok[i1 * T_TOK + s1] = t;
    wgt[i1 * T_TOK + s1] = v1 * inv * SCALE_F;
    int s2 = atomicAdd(&cnt[i2], 1);
    tok[i2 * T_TOK + s2] = t;
    wgt[i2 * T_TOK + s2] = v2 * inv * SCALE_F;
  }
}

// ---------------------------------------------------------------------------
// 2) shared experts (plain store, runs first)
//    512 threads = 16 waves; 32-row M-block; wave-private B slabs in LDS
// ---------------------------------------------------------------------------
__global__ __launch_bounds__(512)
void moe_shared(const float* __restrict__ x,
                const float* __restrict__ wsg, const float* __restrict__ wsu,
                const float* __restrict__ wsd, float* __restrict__ out) {
  __shared__ __bf16   s_x[32][H_DIM + 8];          // 66KB, reused as act
  __shared__ unsigned s_b[16 * 64 * BSTRIDE];      // 80KB: 16 waves x 64 cols

  const int tid = threadIdx.x;
  const int t0  = blockIdx.x * 32;

  for (int i = tid; i < 32 * (H_DIM / 4); i += 512) {
    const int r = i >> 8, c = (i & 255) * 4;
    stage4(&s_x[r][c], x + (size_t)(t0 + r) * H_DIM + c);
  }
  __syncthreads();

  const int wv   = tid >> 5;
  const int lane = tid & 31;
  const int hi   = lane >> 4;
  const int m    = lane & 15;
  const int nlo  = lane & 15;
  const int nb0  = wv * 64;                        // 16 waves * 64 = IS_DIM
  unsigned* sbw  = s_b + wv * 64 * BSTRIDE;        // wave-private slab

  // ---- gate pass ----
  f32x8 accg[2][4] = {};
  for (int kk = 0; kk < H_DIM; kk += 32) {
    stage_b<64>(sbw, wsg + (size_t)kk * IS_DIM + nb0, IS_DIM, lane);
    const bf16x16 a0 = load_a_lds(&s_x[m][kk],      hi);
    const bf16x16 a1 = load_a_lds(&s_x[16 + m][kk], hi);
#pragma unroll
    for (int t4 = 0; t4 < 4; ++t4) {
      const bf16x16 bg = load_b_lds(sbw, t4 * 16 + nlo, hi);
      accg[0][t4] = wmma_bf16(a0, bg, accg[0][t4]);
      accg[1][t4] = wmma_bf16(a1, bg, accg[1][t4]);
    }
  }
  // ---- up pass (reuses same wave-private slab; DS in-order, no barrier) ----
  f32x8 accu[2][4] = {};
  for (int kk = 0; kk < H_DIM; kk += 32) {
    stage_b<64>(sbw, wsu + (size_t)kk * IS_DIM + nb0, IS_DIM, lane);
    const bf16x16 a0 = load_a_lds(&s_x[m][kk],      hi);
    const bf16x16 a1 = load_a_lds(&s_x[16 + m][kk], hi);
#pragma unroll
    for (int t4 = 0; t4 < 4; ++t4) {
      const bf16x16 bu = load_b_lds(sbw, t4 * 16 + nlo, hi);
      accu[0][t4] = wmma_bf16(a0, bu, accu[0][t4]);
      accu[1][t4] = wmma_bf16(a1, bu, accu[1][t4]);
    }
  }
  __syncthreads();                                 // all waves done reading X

  // act = silu(g)*u -> reuse s_x
#pragma unroll
  for (int mi = 0; mi < 2; ++mi)
#pragma unroll
    for (int t4 = 0; t4 < 4; ++t4) {
      const int ncol = nb0 + t4 * 16 + nlo;
#pragma unroll
      for (int r = 0; r < 8; ++r)
        s_x[mi * 16 + r + hi * 8][ncol] =
            (__bf16)(silu_f(accg[mi][t4][r]) * accu[mi][t4][r]);
    }
  __syncthreads();

  // ---- down pass ----
  f32x8 accd[2][4] = {};
  const int nb1 = wv * 64;                         // 16 waves * 64 = H_DIM
  for (int kk = 0; kk < IS_DIM; kk += 32) {
    stage_b<64>(sbw, wsd + (size_t)kk * H_DIM + nb1, H_DIM, lane);
    const bf16x16 a0 = load_a_lds(&s_x[m][kk],      hi);
    const bf16x16 a1 = load_a_lds(&s_x[16 + m][kk], hi);
#pragma unroll
    for (int t4 = 0; t4 < 4; ++t4) {
      const bf16x16 bd = load_b_lds(sbw, t4 * 16 + nlo, hi);
      accd[0][t4] = wmma_bf16(a0, bd, accd[0][t4]);
      accd[1][t4] = wmma_bf16(a1, bd, accd[1][t4]);
    }
  }
#pragma unroll
  for (int mi = 0; mi < 2; ++mi)
#pragma unroll
    for (int t4 = 0; t4 < 4; ++t4) {
      const int ncol = nb1 + t4 * 16 + nlo;
#pragma unroll
      for (int r = 0; r < 8; ++r)
        out[(size_t)(t0 + mi * 16 + r + hi * 8) * H_DIM + ncol] = accd[mi][t4][r];
    }
}

// ---------------------------------------------------------------------------
// 3) routed experts: gather 32 tokens, SwiGLU MLP, weighted atomic combine
//    grid = E_NUM * MT32, 256 threads = 8 waves
// ---------------------------------------------------------------------------
__global__ __launch_bounds__(256)
void moe_routed(const float* __restrict__ x,
                const float* __restrict__ wg, const float* __restrict__ wu,
                const float* __restrict__ wd,
                const int* __restrict__ cnt, const int* __restrict__ tok,
                const float* __restrict__ wgt, float* __restrict__ out) {
  __shared__ __bf16   s_x[32][H_DIM + 8];              // 66KB, reused as act
  __shared__ unsigned s_b[2 * 8 * 64 * BSTRIDE];       // 80KB: g+u slabs / down slab
  __shared__ int      s_tok[32];
  __shared__ float    s_w[32];

  const int e    = blockIdx.x >> 6;                    // / MT32
  const int mt   = blockIdx.x & (MT32 - 1);
  const int nE   = cnt[e];
  const int row0 = mt * 32;
  if (row0 >= nE) return;

  const int tid = threadIdx.x;
  if (tid < 32) {
    const int  r = row0 + tid;
    const bool v = r < nE;
    s_tok[tid] = tok[e * T_TOK + (v ? r : row0)];
    s_w[tid]   = v ? wgt[e * T_TOK + r] : 0.0f;        // padded rows contribute 0
  }
  __syncthreads();

  for (int i = tid; i < 32 * (H_DIM / 4); i += 256) {
    const int r = i >> 8, c = (i & 255) * 4;
    stage4(&s_x[r][c], x + (size_t)s_tok[r] * H_DIM + c);
  }
  __syncthreads();

  const int wv   = tid >> 5;
  const int lane = tid & 31;
  const int hi   = lane >> 4;
  const int m    = lane & 15;
  const int nlo  = lane & 15;

  // ---- gate+up combined pass: per wave 4 N-tiles (8 waves * 64 = I_DIM) ----
  f32x8 accg[2][4] = {}, accu[2][4] = {};
  const int nb0 = wv * 64;
  unsigned* sbg = s_b + wv * 64 * BSTRIDE;
  unsigned* sbu = s_b + (8 * 64 + wv * 64) * BSTRIDE;
  const float* wgE = wg + (size_t)e * H_DIM * I_DIM;
  const float* wuE = wu + (size_t)e * H_DIM * I_DIM;

  for (int kk = 0; kk < H_DIM; kk += 32) {
    stage_b<64>(sbg, wgE + (size_t)kk * I_DIM + nb0, I_DIM, lane);
    stage_b<64>(sbu, wuE + (size_t)kk * I_DIM + nb0, I_DIM, lane);
    const bf16x16 a0 = load_a_lds(&s_x[m][kk],      hi);
    const bf16x16 a1 = load_a_lds(&s_x[16 + m][kk], hi);
#pragma unroll
    for (int t4 = 0; t4 < 4; ++t4) {
      const bf16x16 bg = load_b_lds(sbg, t4 * 16 + nlo, hi);
      accg[0][t4] = wmma_bf16(a0, bg, accg[0][t4]);
      accg[1][t4] = wmma_bf16(a1, bg, accg[1][t4]);
      const bf16x16 bu = load_b_lds(sbu, t4 * 16 + nlo, hi);
      accu[0][t4] = wmma_bf16(a0, bu, accu[0][t4]);
      accu[1][t4] = wmma_bf16(a1, bu, accu[1][t4]);
    }
  }
  __syncthreads();                                     // waves done with X + B slabs

#pragma unroll
  for (int mi = 0; mi < 2; ++mi)
#pragma unroll
    for (int t4 = 0; t4 < 4; ++t4) {
      const int ncol = nb0 + t4 * 16 + nlo;
#pragma unroll
      for (int r = 0; r < 8; ++r)
        s_x[mi * 16 + r + hi * 8][ncol] =
            (__bf16)(silu_f(accg[mi][t4][r]) * accu[mi][t4][r]);
    }
  __syncthreads();

  // ---- down pass: per wave 8 N-tiles (8 waves * 128 = H_DIM), K = I_DIM ----
  f32x8 accd[2][8] = {};
  const int nb1 = wv * 128;
  unsigned* sbd = s_b + wv * 128 * BSTRIDE;
  const float* wdE = wd + (size_t)e * I_DIM * H_DIM;

  for (int kk = 0; kk < I_DIM; kk += 32) {
    stage_b<128>(sbd, wdE + (size_t)kk * H_DIM + nb1, H_DIM, lane);
    const bf16x16 a0 = load_a_lds(&s_x[m][kk],      hi);
    const bf16x16 a1 = load_a_lds(&s_x[16 + m][kk], hi);
#pragma unroll
    for (int t8 = 0; t8 < 8; ++t8) {
      const bf16x16 bd = load_b_lds(sbd, t8 * 16 + nlo, hi);
      accd[0][t8] = wmma_bf16(a0, bd, accd[0][t8]);
      accd[1][t8] = wmma_bf16(a1, bd, accd[1][t8]);
    }
  }

#pragma unroll
  for (int mi = 0; mi < 2; ++mi)
#pragma unroll
    for (int t8 = 0; t8 < 8; ++t8) {
      const int ncol = nb1 + t8 * 16 + nlo;
#pragma unroll
      for (int r = 0; r < 8; ++r) {
        const int mr = mi * 16 + r + hi * 8;
        const float v = accd[mi][t8][r] * s_w[mr];
        unsafeAtomicAdd(&out[(size_t)s_tok[mr] * H_DIM + ncol], v);
      }
    }
}

// ---------------------------------------------------------------------------
extern "C" void kernel_launch(void* const* d_in, const int* in_sizes, int n_in,
                              void* d_out, int out_size, void* d_ws, size_t ws_size,
                              hipStream_t stream) {
  const float* x   = (const float*)d_in[0];
  const float* gw  = (const float*)d_in[1];
  const float* wg  = (const float*)d_in[2];
  const float* wu  = (const float*)d_in[3];
  const float* wd  = (const float*)d_in[4];
  const float* wsg = (const float*)d_in[5];
  const float* wsu = (const float*)d_in[6];
  const float* wsd = (const float*)d_in[7];
  float* out = (float*)d_out;

  char*  ws  = (char*)d_ws;
  int*   cnt = (int*)ws;
  int*   tok = (int*)(ws + 1024);
  float* wgt = (float*)(ws + 1024 + (size_t)E_NUM * T_TOK * 4);

  moe_init  <<<1, 32, 0, stream>>>(cnt);
  moe_router<<<T_TOK / 8, 256, 0, stream>>>(x, gw, cnt, tok, wgt);
  moe_shared<<<T_TOK / 32, 512, 0, stream>>>(x, wsg, wsu, wsd, out);
  moe_routed<<<E_NUM * MT32, 256, 0, stream>>>(x, wg, wu, wd, cnt, tok, wgt, out);
}
